// TruthTransformer_13460427506065
// MI455X (gfx1250) — matrix operand order repaired and checked by reading, not compiled, not run on hardware
//
#include <hip/hip_runtime.h>
#include <hip/hip_bf16.h>

// ---------------- model dims ----------------
enum { BB = 2, SS = 1024, DD = 1024, HH = 16, LL = 3, FFD = 4096, VV = 32000, DKK = 64 };

typedef __bf16 bf16_t;
typedef bf16_t v16bf __attribute__((ext_vector_type(16)));
typedef bf16_t bf16x8 __attribute__((ext_vector_type(8)));
typedef bf16_t bf16x4 __attribute__((ext_vector_type(4)));
typedef float  v8f   __attribute__((ext_vector_type(8)));

union FragBF16 { v16bf v; bf16x8 h[2]; };

static __device__ __forceinline__ bf16x4 pack4(float4 a) {
    bf16x4 r;
    r[0] = (bf16_t)a.x; r[1] = (bf16_t)a.y; r[2] = (bf16_t)a.z; r[3] = (bf16_t)a.w;
    return r;
}

// ---------------------------------------------------------------------------
// Generic batched GEMM: C = alpha * (A @ B[^T]) + bias, optional ReLU.
// A: [M,K] fp32 (row stride lda), B: [K,N] fp32 (row stride ldb, or [N,K] if
// transB), C: [M,N] fp32 (row stride ldc). Two-level batch: z = zo*innerN+zi.
// Internally bf16 WMMA (v_wmma_f32_16x16x32_bf16) with f32 accumulate.
// Workgroup tile: 256x128, 256 thr = 8 waves in a 4x2 grid; each wave owns a
// 64x64 sub-tile = 4x4 WMMA tiles (16 wmma : 16 ds_load_b128 per K-step).
// LDS double-buffered (60 KB of the 320 KB WGP pool): one barrier per K-step.
// Assumptions (all launches here): M % 256 == 0, K % 32 == 0, N % 16 == 0.
// ---------------------------------------------------------------------------
#define TM 256
#define TN 128
#define TK 32
#define APAD 8   // pad rows to 40 bf16 (80B): rotates LDS banks, keeps 16B align
#define BPAD 8

__global__ __launch_bounds__(256) void gemm_bf16_wmma(
    const float* __restrict__ A, const float* __restrict__ Bm,
    const float* __restrict__ bias, float* __restrict__ C,
    int M, int N, int K,
    long long lda, long long ldb, long long ldc,
    int innerN,
    long long sAin, long long sAout,
    long long sBin, long long sBout,
    long long sCin, long long sCout,
    float alpha, int flags)
{
    const int z  = blockIdx.z;
    const int zo = z / innerN, zi = z % innerN;
    A  += (long long)zo * sAout + (long long)zi * sAin;
    Bm += (long long)zo * sBout + (long long)zi * sBin;
    C  += (long long)zo * sCout + (long long)zi * sCin;

    const bool transB = (flags & 1) != 0;
    const bool relu   = (flags & 2) != 0;

    __shared__ __align__(16) bf16_t Al [2][TM][TK + APAD];  // row-major A tiles
    __shared__ __align__(16) bf16_t Blt[2][TN][TK + BPAD];  // B tiles, n-major

    const int tid  = threadIdx.x;
    const int lane = tid & 31;
    const int wave = tid >> 5;
    const int wrow = wave >> 1;   // 0..3  -> rows wrow*64
    const int wcol = wave & 1;    // 0..1  -> cols wcol*64
    const int bm   = blockIdx.y * TM;
    const int bn   = blockIdx.x * TN;

    // staging coordinates (fixed per thread)
    const int ar0 = tid >> 3;             // base row 0..31 (8 passes of 32 rows)
    const int ak0 = (tid & 7) * 4;        // 0..28 : lane's 16B chunk of a row
    const int bn0 = (tid & 31) * 4;       // 0..124
    const int bk0 = (tid >> 5) * 4;       // 0..28

    v8f acc[4][4];
    for (int i = 0; i < 4; ++i)
        for (int j = 0; j < 4; ++j)
            for (int r = 0; r < 8; ++r) acc[i][j][r] = 0.0f;

    // ---- tile stager: 8 lanes cover one A row's 128B contiguously
    auto stage = [&](int sb, int kt) {
        {   // A tile: TM x TK, 8 passes of 32 rows; 1 float4 + 1 b64 store per pass
            float4 f[8];
            for (int p = 0; p < 8; ++p) {
                int row = ar0 + p * 32;
                f[p] = *(const float4*)&A[(long long)(bm + row) * lda + kt + ak0];
            }
            for (int p = 0; p < 8; ++p)
                *(bf16x4*)&Al[sb][ar0 + p * 32][ak0] = pack4(f[p]);
        }
        {   // B tile: TK x TN into n-major LDS; thread owns 4k x 4n
            float vals[4][4];   // [kk][nn]
            if (!transB) {
                if (bn + bn0 < N) {    // whole float4 in range (N % 4 == 0)
                    for (int kk = 0; kk < 4; ++kk) {
                        float4 f = *(const float4*)&Bm[(long long)(kt + bk0 + kk) * ldb + bn + bn0];
                        vals[kk][0] = f.x; vals[kk][1] = f.y; vals[kk][2] = f.z; vals[kk][3] = f.w;
                    }
                } else {
                    for (int kk = 0; kk < 4; ++kk)
                        for (int nn = 0; nn < 4; ++nn) vals[kk][nn] = 0.0f;
                }
            } else {
                for (int nn = 0; nn < 4; ++nn) {
                    int gn = bn + bn0 + nn;
                    int gc = gn < N - 1 ? gn : N - 1;                     // clamp
                    float4 f = *(const float4*)&Bm[(long long)gc * ldb + kt + bk0];
                    bool ok = gn < N;                                     // select
                    vals[0][nn] = ok ? f.x : 0.0f;
                    vals[1][nn] = ok ? f.y : 0.0f;
                    vals[2][nn] = ok ? f.z : 0.0f;
                    vals[3][nn] = ok ? f.w : 0.0f;
                }
            }
            for (int nn = 0; nn < 4; ++nn) {
                bf16x4 p;
                p[0] = (bf16_t)vals[0][nn]; p[1] = (bf16_t)vals[1][nn];
                p[2] = (bf16_t)vals[2][nn]; p[3] = (bf16_t)vals[3][nn];
                *(bf16x4*)&Blt[sb][bn0 + nn][bk0] = p;    // 8B ds_store
            }
        }
    };

    stage(0, 0);
    __syncthreads();

    const int kb = (lane >> 4) * 8;    // A frag K base (ISA 7.12.2)
    const int k0 = (lane >> 4) * 8;    // B frag K base

    int buf = 0;
    for (int kt = 0; kt < K; kt += TK, buf ^= 1) {
        // stage next tile into the other buffer (overlaps with WMMA below)
        if (kt + TK < K) {
            stage(buf ^ 1, kt + TK);
            if (kt + 2 * TK < K)   // prefetch K-tile after next (global_prefetch_b8)
                __builtin_prefetch(&A[(long long)(bm + ar0) * lda + kt + 2 * TK + ak0], 0, 0);
        }

        // ---- fragments per ISA 7.12.2 layouts
        // B 32x16 bf16: lane l -> n = l&15; k = (l>>4)*8 + (0..7), then +16 + (0..7)
        FragBF16 bf[4];
        for (int ct = 0; ct < 4; ++ct) {
            int nl = wcol * 64 + ct * 16 + (lane & 15);
            bf[ct].h[0] = *(const bf16x8*)&Blt[buf][nl][k0];
            bf[ct].h[1] = *(const bf16x8*)&Blt[buf][nl][k0 + 16];
        }
        // A 16x32 bf16: lane l -> m = l&15, kbase = (l>>4)*8;
        //   VGPR0-3: k = kbase..kbase+7, VGPR4-7: k = kbase+16..kbase+23
        for (int rt = 0; rt < 4; ++rt) {
            FragBF16 af;
            int row = wrow * 64 + rt * 16 + (lane & 15);
            af.h[0] = *(const bf16x8*)&Al[buf][row][kb];
            af.h[1] = *(const bf16x8*)&Al[buf][row][kb + 16];
            for (int ct = 0; ct < 4; ++ct)
                acc[rt][ct] = __builtin_amdgcn_wmma_f32_16x16x32_bf16(
                    false, af.v, false, bf[ct].v,
                    (short)0, acc[rt][ct], false, false);
        }
        __syncthreads();   // single barrier: protects both buffers' reuse edges
    }

    // ---- store: C/D layout: lane l -> n = l&15; VGPR r -> m = (l>>4)*8 + r
    // (M tiles always full; N guard is wave-uniform since N % 16 == 0)
    for (int rt = 0; rt < 4; ++rt) {
        int mbase = bm + wrow * 64 + rt * 16 + (lane >> 4) * 8;
        for (int ct = 0; ct < 4; ++ct) {
            int n = bn + wcol * 64 + ct * 16 + (lane & 15);
            if (n >= N) continue;
            float bv = bias ? bias[n] : 0.0f;
            for (int r = 0; r < 8; ++r) {
                float xv = acc[rt][ct][r] * alpha + bv;
                if (relu) xv = xv > 0.0f ? xv : 0.0f;
                C[(long long)(mbase + r) * ldc + n] = xv;
            }
        }
    }
}

// ---------------------------------------------------------------------------
// h[b,s,:] = emb[x[b,s],:] + 0.05 * noise[b,s,:]
// ---------------------------------------------------------------------------
__global__ __launch_bounds__(256) void embed_noise(
    const int* __restrict__ x, const float* __restrict__ noise,
    const float* __restrict__ emb, float* __restrict__ h)
{
    long long i = (long long)blockIdx.x * 256 + threadIdx.x;  // over B*S*D
    long long sd = i >> 10;             // token index within [B*S]
    int d = (int)(i & (DD - 1));
    int tok = x[sd];
    h[i] = emb[(long long)tok * DD + d] + 0.05f * noise[i];
}

// ---------------------------------------------------------------------------
// Row softmax over 1024 columns (in place). One 256-thread block per row.
// NOTE: the reference's "empathy bias" 0.5*cos[b,h] is constant per softmax
// row -> shift-invariance makes it a mathematical no-op; it is omitted.
// ---------------------------------------------------------------------------
__global__ __launch_bounds__(256) void softmax_rows(float* __restrict__ p)
{
    __shared__ float red[256];
    long long row = blockIdx.x;
    float* xr = p + row * (long long)SS;
    int t = threadIdx.x;
    float v[4];
    float m = -1e30f;
    for (int i = 0; i < 4; ++i) { v[i] = xr[t + i * 256]; m = fmaxf(m, v[i]); }
    red[t] = m; __syncthreads();
    for (int s = 128; s > 0; s >>= 1) { if (t < s) red[t] = fmaxf(red[t], red[t + s]); __syncthreads(); }
    m = red[0]; __syncthreads();
    float sum = 0.0f;
    for (int i = 0; i < 4; ++i) { v[i] = __expf(v[i] - m); sum += v[i]; }
    red[t] = sum; __syncthreads();
    for (int s = 128; s > 0; s >>= 1) { if (t < s) red[t] += red[t + s]; __syncthreads(); }
    float inv = 1.0f / red[0];
    for (int i = 0; i < 4; ++i) xr[t + i * 256] = v[i] * inv;
}

// ---------------------------------------------------------------------------
// h[row,:] = LayerNorm(h[row,:] + resid[row,:]) * g + b   (D = 1024)
// ---------------------------------------------------------------------------
__global__ __launch_bounds__(256) void add_layernorm(
    float* __restrict__ h, const float* __restrict__ resid,
    const float* __restrict__ g, const float* __restrict__ b)
{
    __shared__ float red[256];
    long long row = blockIdx.x;
    float* xr = h + row * (long long)DD;
    const float* rr = resid + row * (long long)DD;
    int t = threadIdx.x;
    float v[4]; float s = 0.0f;
    for (int i = 0; i < 4; ++i) { int d = t + i * 256; v[i] = xr[d] + rr[d]; s += v[i]; }
    red[t] = s; __syncthreads();
    for (int st = 128; st > 0; st >>= 1) { if (t < st) red[t] += red[t + st]; __syncthreads(); }
    float mean = red[0] * (1.0f / DD); __syncthreads();
    float d2 = 0.0f;
    for (int i = 0; i < 4; ++i) { float dd = v[i] - mean; d2 += dd * dd; }
    red[t] = d2; __syncthreads();
    for (int st = 128; st > 0; st >>= 1) { if (t < st) red[t] += red[t + st]; __syncthreads(); }
    float rstd = rsqrtf(red[0] * (1.0f / DD) + 1e-5f);
    for (int i = 0; i < 4; ++i) {
        int d = t + i * 256;
        xr[d] = (v[i] - mean) * rstd * g[d] + b[d];
    }
}

// ---------------------------------------------------------------------------
static inline void launch_gemm(hipStream_t s,
    const float* A, const float* Bm, const float* bias, float* C,
    int M, int N, int K, long long lda, long long ldb, long long ldc,
    int batch, int innerN,
    long long sAin, long long sAout, long long sBin, long long sBout,
    long long sCin, long long sCout, float alpha, int flags)
{
    dim3 grid((N + TN - 1) / TN, (M + TM - 1) / TM, batch);
    gemm_bf16_wmma<<<grid, 256, 0, s>>>(A, Bm, bias, C, M, N, K, lda, ldb, ldc,
                                        innerN, sAin, sAout, sBin, sBout,
                                        sCin, sCout, alpha, flags);
}

extern "C" void kernel_launch(void* const* d_in, const int* in_sizes, int n_in,
                              void* d_out, int out_size, void* d_ws, size_t ws_size,
                              hipStream_t stream)
{
    const int*   x     = (const int*)  d_in[0];
    const float* noise = (const float*)d_in[1];
    const float* emb   = (const float*)d_in[2];
    const float* Wq    = (const float*)d_in[3];  const float* bq   = (const float*)d_in[4];
    const float* Wk    = (const float*)d_in[5];  const float* bk   = (const float*)d_in[6];
    const float* Wv    = (const float*)d_in[7];  const float* bv   = (const float*)d_in[8];
    const float* Wo    = (const float*)d_in[9];  const float* bo   = (const float*)d_in[10];
    const float* W1f   = (const float*)d_in[11]; const float* b1f  = (const float*)d_in[12];
    const float* W2f   = (const float*)d_in[13]; const float* b2f  = (const float*)d_in[14];
    const float* g1    = (const float*)d_in[15]; const float* be1  = (const float*)d_in[16];
    const float* g2    = (const float*)d_in[17]; const float* be2  = (const float*)d_in[18];
    const float* Wout  = (const float*)d_in[19]; const float* bout = (const float*)d_in[20];
    float* out = (float*)d_out;

    const int BS = BB * SS;                 // 2048
    // ---- workspace carve (fp32 buffers); ~200 MB total
    char* ws = (char*)d_ws;
    float* h  = (float*)(ws + ((size_t)0  << 20));   //  8 MB  [B,S,D]
    float* q  = (float*)(ws + ((size_t)8  << 20));   //  8 MB  [B,S,D]
    float* k  = (float*)(ws + ((size_t)16 << 20));   //  8 MB  [B,S,D]
    float* v  = (float*)(ws + ((size_t)24 << 20));   //  8 MB  [B,S,D]
    float* t0 = (float*)(ws + ((size_t)32 << 20));   //  8 MB  [B,S,D]
    float* ff = (float*)(ws + ((size_t)40 << 20));   // 32 MB  [B,S,FF]
    float* sc = (float*)(ws + ((size_t)72 << 20));   // 128 MB [B,H,S,S] (L2-resident)

    // ---- embedding + noise
    {
        long long n = (long long)BS * DD;
        embed_noise<<<dim3((unsigned)(n / 256)), 256, 0, stream>>>(x, noise, emb, h);
    }

    const long long SD  = (long long)SS * DD;
    const long long SSQ = (long long)SS * SS;

    for (int l = 0; l < LL; ++l) {
        const float* Wq_l = Wq + (long long)l * DD * DD;  const float* bq_l = bq + (long long)l * DD;
        const float* Wk_l = Wk + (long long)l * DD * DD;  const float* bk_l = bk + (long long)l * DD;
        const float* Wv_l = Wv + (long long)l * DD * DD;  const float* bv_l = bv + (long long)l * DD;
        const float* Wo_l = Wo + (long long)l * DD * DD;  const float* bo_l = bo + (long long)l * DD;
        const float* W1_l = W1f + (long long)l * DD * FFD; const float* b1_l = b1f + (long long)l * FFD;
        const float* W2_l = W2f + (long long)l * FFD * DD; const float* b2_l = b2f + (long long)l * DD;

        // Q/K/V projections: [BS,D] = h @ W + b
        launch_gemm(stream, h, Wq_l, bq_l, q, BS, DD, DD, DD, DD, DD,
                    1, 1, 0,0, 0,0, 0,0, 1.0f, 0);
        launch_gemm(stream, h, Wk_l, bk_l, k, BS, DD, DD, DD, DD, DD,
                    1, 1, 0,0, 0,0, 0,0, 1.0f, 0);
        launch_gemm(stream, h, Wv_l, bv_l, v, BS, DD, DD, DD, DD, DD,
                    1, 1, 0,0, 0,0, 0,0, 1.0f, 0);

        // scores[b,h] = (1/8) * Q[b,:,h,:] @ K[b,:,h,:]^T   (batch = B*H = 32)
        launch_gemm(stream, q, k, nullptr, sc,
                    SS, SS, DKK, DD, DD, SS,
                    BB * HH, HH,
                    /*sAin*/ DKK, /*sAout*/ SD,
                    /*sBin*/ DKK, /*sBout*/ SD,
                    /*sCin*/ SSQ, /*sCout*/ (long long)HH * SSQ,
                    0.125f, /*transB*/ 1);

        // softmax over last dim (empathy cos bias dropped: per-row constant)
        softmax_rows<<<dim3((unsigned)(BB * HH * SS)), 256, 0, stream>>>(sc);

        // attn[b,:,h,:] = P[b,h] @ V[b,:,h,:]   -> t0 in [B,S,D] layout
        launch_gemm(stream, sc, v, nullptr, t0,
                    SS, DKK, SS, SS, DD, DD,
                    BB * HH, HH,
                    /*sAin*/ SSQ, /*sAout*/ (long long)HH * SSQ,
                    /*sBin*/ DKK, /*sBout*/ SD,
                    /*sCin*/ DKK, /*sCout*/ SD,
                    1.0f, 0);

        // output projection into q (free), then h = LN(h + q)
        launch_gemm(stream, t0, Wo_l, bo_l, q, BS, DD, DD, DD, DD, DD,
                    1, 1, 0,0, 0,0, 0,0, 1.0f, 0);
        add_layernorm<<<dim3((unsigned)BS), 256, 0, stream>>>(h, q, g1 + (long long)l * DD, be1 + (long long)l * DD);

        // FFN: ff = relu(h @ W1 + b1); t0 = ff @ W2 + b2; h = LN(h + t0)
        launch_gemm(stream, h, W1_l, b1_l, ff, BS, FFD, DD, DD, FFD, FFD,
                    1, 1, 0,0, 0,0, 0,0, 1.0f, /*relu*/ 2);
        launch_gemm(stream, ff, W2_l, b2_l, t0, BS, DD, FFD, FFD, DD, DD,
                    1, 1, 0,0, 0,0, 0,0, 1.0f, 0);
        add_layernorm<<<dim3((unsigned)BS), 256, 0, stream>>>(h, t0, g2 + (long long)l * DD, be2 + (long long)l * DD);
    }

    // logits: out[BS, V] = h @ Wout + bout
    launch_gemm(stream, h, Wout, bout, out, BS, VV, DD, DD, VV, VV,
                1, 1, 0,0, 0,0, 0,0, 1.0f, 0);
}
